// OxTM_70360154243185
// MI455X (gfx1250) — compile-verified, matrix operand order checked by reading, not run on hardware
//
#include <hip/hip_runtime.h>
#include <hip/hip_bf16.h>
#include <math.h>

// ---------------------------------------------------------------------------
// Types for CDNA5 WMMA (gfx1250, wave32)
// ---------------------------------------------------------------------------
typedef __bf16 v8bf  __attribute__((ext_vector_type(8)));
typedef __bf16 v16bf __attribute__((ext_vector_type(16)));
typedef float  v8f   __attribute__((ext_vector_type(8)));

// ---------------------------------------------------------------------------
// Generic GEMM: C[M,N] = act(A[M,K] @ B[K,N] + bias[N])
// f32 in/out, bf16 WMMA (v_wmma_f32_16x16x32_bf16) internally, f32 accum.
// 64x64 output tile per 256-thread (8-wave) workgroup; K-step 32.
//   wave w: row-tile = w>>1 (16 rows), col-tiles = (w&1)*2 + {0,1}.
// A staged row-major in LDS, B staged TRANSPOSED ([n][k]) so both fragment
// loads are two aligned 16B LDS vector reads per operand.
// CDNA5 16-bit A layout: lane<16 -> K {0-7,16-23}, lane>=16 -> K {8-15,24-31};
// same K pattern for B with col = lane&15.
// Interior tiles take a vectorized staging path (2x b128 global loads +
// ds_store_b128 / ds_store_b16), edge tiles fall back to bounds-checked code.
// act: 0 = none, 1 = softplus
// ---------------------------------------------------------------------------
__global__ __launch_bounds__(256)
void gemm_bf16_wmma(const float* __restrict__ A, const float* __restrict__ B,
                    const float* __restrict__ bias, float* __restrict__ C,
                    int M, int N, int K, int act)
{
    __shared__ v8bf sA[64][5];   // 64 rows x 40 bf16 (32 used + pad, 16B chunks)
    __shared__ v8bf sB[64][5];   // 64 cols x 40 bf16 (transposed: [n][k])
    __bf16* sAe = (__bf16*)sA;
    __bf16* sBe = (__bf16*)sB;

    const int tid   = threadIdx.x;
    const int wave  = tid >> 5;
    const int lane  = tid & 31;
    const int l16   = lane & 15;
    const int hsel  = lane >> 4;        // 0 -> K{0-7,16-23}, 1 -> K{8-15,24-31}
    const int rtile = wave >> 1;        // 0..3
    const int c0    = (wave & 1) * 2;   // 0 or 2

    const int m0 = blockIdx.x * 64;
    const int n0 = blockIdx.y * 64;

    // Fast-path decomposition (one tile element group per thread, no loops):
    //   A: thread t -> row m = t>>2, k-group (t&3)*8  (8 consecutive K)
    //   B: thread t -> k-row  t>>3,  n-group (t&7)*8  (8 consecutive N)
    const int am = tid >> 2, akg = (tid & 3) << 3;
    const int bk = tid >> 3, bng = (tid & 7) << 3;

    const bool aVec = (m0 + 64 <= M) && ((K & 3) == 0);
    const bool bVec = (n0 + 64 <= N) && ((N & 3) == 0);

    v8f acc0 = {};
    v8f acc1 = {};

    const int Kt = (K + 31) >> 5;
    for (int kt = 0; kt < Kt; ++kt) {
        const int k0 = kt << 5;
        const bool interiorK = (k0 + 32 <= K);

        // ---- Stage A tile (64x32) f32 -> bf16 ----
        if (aVec && interiorK) {
            const float4* src = (const float4*)(A + (size_t)(m0 + am) * K + k0 + akg);
            float4 p0 = src[0];
            float4 p1 = src[1];
            v8bf v;
            v[0] = (__bf16)p0.x; v[1] = (__bf16)p0.y;
            v[2] = (__bf16)p0.z; v[3] = (__bf16)p0.w;
            v[4] = (__bf16)p1.x; v[5] = (__bf16)p1.y;
            v[6] = (__bf16)p1.z; v[7] = (__bf16)p1.w;
            *(v8bf*)(sAe + am * 40 + akg) = v;
        } else {
            for (int f = tid; f < 64 * 32; f += 256) {
                int m = f >> 5, kk = f & 31;
                int gm = m0 + m, gk = k0 + kk;
                float v = (gm < M && gk < K) ? A[(size_t)gm * K + gk] : 0.f;
                sAe[m * 40 + kk] = (__bf16)v;
            }
        }

        // ---- Stage B tile (32x64) transposed into [n][k] ----
        if (bVec && interiorK) {
            const float4* src = (const float4*)(B + (size_t)(k0 + bk) * N + n0 + bng);
            float4 p0 = src[0];
            float4 p1 = src[1];
            __bf16 b[8];
            b[0] = (__bf16)p0.x; b[1] = (__bf16)p0.y;
            b[2] = (__bf16)p0.z; b[3] = (__bf16)p0.w;
            b[4] = (__bf16)p1.x; b[5] = (__bf16)p1.y;
            b[6] = (__bf16)p1.z; b[7] = (__bf16)p1.w;
            #pragma unroll
            for (int q = 0; q < 8; ++q)
                sBe[(bng + q) * 40 + bk] = b[q];
        } else {
            for (int f = tid; f < 64 * 32; f += 256) {
                int kk = f >> 6, nn = f & 63;
                int gk = k0 + kk, gn = n0 + nn;
                float v = (gk < K && gn < N) ? B[(size_t)gk * N + gn] : 0.f;
                sBe[nn * 40 + kk] = (__bf16)v;
            }
        }

        // CDNA5 prefetch of next K-slab (global_prefetch_b8).
        if (k0 + 32 < K) {
            int pm = m0 + (tid & 63);
            if (pm < M)
                __builtin_prefetch(&A[(size_t)pm * K + k0 + 32], 0, 1);
            if (k0 + 63 < K)
                __builtin_prefetch(&B[(size_t)(k0 + 32 + (tid & 31)) * N + n0], 0, 1);
        }
        __syncthreads();

        // ---- Fragments: two aligned 16B LDS loads per operand ----
        v8bf alo = sA[rtile * 16 + l16][hsel];
        v8bf ahi = sA[rtile * 16 + l16][hsel + 2];
        v16bf af = __builtin_shufflevector(alo, ahi,
                     0,1,2,3,4,5,6,7,8,9,10,11,12,13,14,15);

        v8bf b0lo = sB[(c0 + 0) * 16 + l16][hsel];
        v8bf b0hi = sB[(c0 + 0) * 16 + l16][hsel + 2];
        v16bf bf0 = __builtin_shufflevector(b0lo, b0hi,
                     0,1,2,3,4,5,6,7,8,9,10,11,12,13,14,15);

        v8bf b1lo = sB[(c0 + 1) * 16 + l16][hsel];
        v8bf b1hi = sB[(c0 + 1) * 16 + l16][hsel + 2];
        v16bf bf1 = __builtin_shufflevector(b1lo, b1hi,
                     0,1,2,3,4,5,6,7,8,9,10,11,12,13,14,15);

        acc0 = __builtin_amdgcn_wmma_f32_16x16x32_bf16(false, af, false, bf0,
                                                       (short)0, acc0, false, false);
        acc1 = __builtin_amdgcn_wmma_f32_16x16x32_bf16(false, af, false, bf1,
                                                       (short)0, acc1, false, false);
        __syncthreads();
    }

    // Epilogue: CDNA5 f32 C/D layout -> element v is row (v + hsel*8), col lane&15.
    const int mbase = m0 + rtile * 16 + hsel * 8;
    {
        int gn = n0 + (c0 + 0) * 16 + l16;
        if (gn < N) {
            float bv = bias ? bias[gn] : 0.f;
            #pragma unroll
            for (int v = 0; v < 8; ++v) {
                int gm = mbase + v;
                if (gm < M) {
                    float val = acc0[v] + bv;
                    if (act == 1) val = (val > 20.f) ? val : log1pf(__expf(val));
                    C[(size_t)gm * N + gn] = val;
                }
            }
        }
    }
    {
        int gn = n0 + (c0 + 1) * 16 + l16;
        if (gn < N) {
            float bv = bias ? bias[gn] : 0.f;
            #pragma unroll
            for (int v = 0; v < 8; ++v) {
                int gm = mbase + v;
                if (gm < M) {
                    float val = acc1[v] + bv;
                    if (act == 1) val = (val > 20.f) ? val : log1pf(__expf(val));
                    C[(size_t)gm * N + gn] = val;
                }
            }
        }
    }
}

// ---------------------------------------------------------------------------
// Column mean / biased variance over M rows (thread per column, coalesced).
// ---------------------------------------------------------------------------
__global__ __launch_bounds__(256)
void colstats(const float* __restrict__ Y, float* __restrict__ mean,
              float* __restrict__ var, int M, int N)
{
    int j = blockIdx.x * 256 + threadIdx.x;
    if (j >= N) return;
    float s = 0.f, s2 = 0.f;
    for (int i = 0; i < M; ++i) {
        float y = Y[(size_t)i * N + j];
        s += y; s2 += y * y;
    }
    float m = s / (float)M;
    mean[j] = m;
    var[j]  = s2 / (float)M - m * m;
}

// ---------------------------------------------------------------------------
// Per-row: batchnorm(mu_pre), batchnorm(lv_pre), reparameterize, softmax(z)
// -> theta; also accumulate the KL divergence. One block of 128 per row, K=100.
// ---------------------------------------------------------------------------
__global__ __launch_bounds__(128)
void bn_reparam_theta(const float* __restrict__ mu_pre, const float* __restrict__ lv_pre,
                      const float* __restrict__ mm, const float* __restrict__ mv,
                      const float* __restrict__ lm, const float* __restrict__ lvv,
                      const float* __restrict__ eps, float* __restrict__ theta,
                      float* kld_acc)
{
    const int K = 100;
    __shared__ float z[128];
    __shared__ float red[128];
    int i = blockIdx.x, t = threadIdx.x;

    float zt = -__builtin_inff();
    float kterm = 0.f;
    if (t < K) {
        float mp  = mu_pre[(size_t)i * K + t];
        float lp  = lv_pre[(size_t)i * K + t];
        float mun = (mp - mm[t]) * rsqrtf(mv[t] + 1e-5f);
        float lvn = (lp - lm[t]) * rsqrtf(lvv[t] + 1e-5f);
        zt = mun + __expf(0.5f * lvn) * eps[(size_t)i * K + t];
        const float var2 = 1.f - 1.f / 100.f;
        float var = __expf(lvn);
        kterm = var / var2 + mun * mun / var2 + logf(var2) - lvn;
    }
    z[t] = zt;
    red[t] = kterm;
    __syncthreads();
    for (int s = 64; s > 0; s >>= 1) { if (t < s) red[t] += red[t + s]; __syncthreads(); }
    if (t == 0) atomicAdd(kld_acc, 0.5f * (red[0] - (float)K));
    __syncthreads();

    red[t] = z[t];
    __syncthreads();
    for (int s = 64; s > 0; s >>= 1) { if (t < s) red[t] = fmaxf(red[t], red[t + s]); __syncthreads(); }
    float zmax = red[0];
    __syncthreads();
    float e = (t < K) ? __expf(z[t] - zmax) : 0.f;
    red[t] = e;
    __syncthreads();
    for (int s = 64; s > 0; s >>= 1) { if (t < s) red[t] += red[t + s]; __syncthreads(); }
    float Z = red[0];
    if (t < K) theta[(size_t)i * K + t] = e / Z;
}

// ---------------------------------------------------------------------------
// Per-row (block per row): batchnorm with column stats, softmax over N,
// recon loss -sum_j x*log(p + 1e-10). Three streaming sweeps (L2-resident).
// ---------------------------------------------------------------------------
__global__ __launch_bounds__(256)
void row_softmax_loss(const float* __restrict__ Y, const float* __restrict__ cm,
                      const float* __restrict__ cv, const float* __restrict__ X,
                      float* recon_acc, int N)
{
    __shared__ float red[256];
    int i = blockIdx.x, t = threadIdx.x;
    const size_t row = (size_t)i * N;

    float lmax = -__builtin_inff();
    for (int j = t; j < N; j += 256) {
        float tn = (Y[row + j] - cm[j]) * rsqrtf(cv[j] + 1e-5f);
        lmax = fmaxf(lmax, tn);
    }
    red[t] = lmax; __syncthreads();
    for (int s = 128; s > 0; s >>= 1) { if (t < s) red[t] = fmaxf(red[t], red[t + s]); __syncthreads(); }
    float rmax = red[0]; __syncthreads();

    float lsum = 0.f;
    for (int j = t; j < N; j += 256) {
        float tn = (Y[row + j] - cm[j]) * rsqrtf(cv[j] + 1e-5f);
        lsum += __expf(tn - rmax);
    }
    red[t] = lsum; __syncthreads();
    for (int s = 128; s > 0; s >>= 1) { if (t < s) red[t] += red[t + s]; __syncthreads(); }
    float Z = red[0]; __syncthreads();
    float invZ = 1.f / Z;

    float ll = 0.f;
    for (int j = t; j < N; j += 256) {
        float tn = (Y[row + j] - cm[j]) * rsqrtf(cv[j] + 1e-5f);
        float p  = __expf(tn - rmax) * invZ;
        ll += X[row + j] * logf(p + 1e-10f);
    }
    red[t] = ll; __syncthreads();
    for (int s = 128; s > 0; s >>= 1) { if (t < s) red[t] += red[t + s]; __syncthreads(); }
    if (t == 0) atomicAdd(recon_acc, -red[0]);
}

// ---------------------------------------------------------------------------
// L2-normalize rows of [rows, D]. Block (64 threads) per row.
// ---------------------------------------------------------------------------
__global__ __launch_bounds__(64)
void norm_rows(const float* __restrict__ X, float* __restrict__ out, int D)
{
    __shared__ float red[64];
    int r = blockIdx.x, t = threadIdx.x;
    float s = 0.f;
    for (int k = t; k < D; k += 64) { float v = X[(size_t)r * D + k]; s += v * v; }
    red[t] = s; __syncthreads();
    for (int st = 32; st > 0; st >>= 1) { if (t < st) red[t] += red[t + st]; __syncthreads(); }
    float rn = rsqrtf(red[0]);
    for (int k = t; k < D; k += 64) out[(size_t)r * D + k] = X[(size_t)r * D + k] * rn;
}

// ---------------------------------------------------------------------------
// Top-100 of each 20000-wide row via 100 masked argmax passes. Block per topic.
// ---------------------------------------------------------------------------
__global__ __launch_bounds__(256)
void topk100(const float* __restrict__ beta, float* __restrict__ scratch,
             float* __restrict__ vals, int* __restrict__ idxs, int V)
{
    __shared__ float rv[256];
    __shared__ int   ri[256];
    int topic = blockIdx.x, t = threadIdx.x;
    float* s = scratch + (size_t)topic * V;
    for (int j = t; j < V; j += 256) s[j] = beta[(size_t)topic * V + j];
    __syncthreads();
    for (int p = 0; p < 100; ++p) {
        float bv = -__builtin_inff(); int bi = 0x7fffffff;
        for (int j = t; j < V; j += 256) {
            float v = s[j];
            if (v > bv || (v == bv && j < bi)) { bv = v; bi = j; }
        }
        rv[t] = bv; ri[t] = bi; __syncthreads();
        for (int st = 128; st > 0; st >>= 1) {
            if (t < st) {
                if (rv[t + st] > rv[t] || (rv[t + st] == rv[t] && ri[t + st] < ri[t])) {
                    rv[t] = rv[t + st]; ri[t] = ri[t + st];
                }
            }
            __syncthreads();
        }
        if (t == 0) {
            vals[topic * 100 + p] = rv[0];
            idxs[topic * 100 + p] = ri[0];
            s[ri[0]] = -__builtin_inff();
        }
        __syncthreads();
    }
}

// ---------------------------------------------------------------------------
// Per-topic 100x100 cost tile: C[i][j] = 1 - en_n[icn[i]] . cn_n[ien[j]]
// (matches reference's M[cn_idx][:, en_idx] indexing). Never builds 20000^2 M.
// ---------------------------------------------------------------------------
__global__ __launch_bounds__(256)
void build_cost(const float* __restrict__ en_n, const float* __restrict__ cn_n,
                const int* __restrict__ icn, const int* __restrict__ ien,
                float* __restrict__ Cm)
{
    int topic = blockIdx.x;
    for (int p = threadIdx.x; p < 10000; p += 256) {
        int i = p / 100, j = p % 100;
        const float* er = en_n + (size_t)icn[topic * 100 + i] * 300;
        const float* cr = cn_n + (size_t)ien[topic * 100 + j] * 300;
        float d = 0.f;
        for (int k = 0; k < 300; ++k) d += er[k] * cr[k];
        Cm[(size_t)topic * 10000 + p] = 1.f - d;
    }
}

// ---------------------------------------------------------------------------
// Log-domain Sinkhorn, 1000 iterations, block per topic, C in LDS (40KB).
// Online log-sum-exp per row/col (reg = 1e-3 -> must stay in log domain).
// ---------------------------------------------------------------------------
__global__ __launch_bounds__(128)
void sinkhorn100(const float* __restrict__ Cmat, const float* __restrict__ vcn,
                 const float* __restrict__ ven, float* sink_acc)
{
    __shared__ float C[10000];
    __shared__ float f[100], g[100], la[100], lb[100];
    __shared__ float red[128];
    const float reg = 0.001f, invreg = 1000.f;
    int topic = blockIdx.x, t = threadIdx.x;
    const float* Cg = Cmat + (size_t)topic * 10000;
    for (int p = t; p < 10000; p += 128) C[p] = Cg[p];

    red[t] = (t < 100) ? vcn[topic * 100 + t] : 0.f;
    __syncthreads();
    for (int s = 64; s > 0; s >>= 1) { if (t < s) red[t] += red[t + s]; __syncthreads(); }
    float sa = red[0]; __syncthreads();
    if (t < 100) la[t] = __logf(vcn[topic * 100 + t]) - __logf(sa);

    red[t] = (t < 100) ? ven[topic * 100 + t] : 0.f;
    __syncthreads();
    for (int s = 64; s > 0; s >>= 1) { if (t < s) red[t] += red[t + s]; __syncthreads(); }
    float sb = red[0];
    if (t < 100) { lb[t] = __logf(ven[topic * 100 + t]) - __logf(sb); f[t] = 0.f; g[t] = 0.f; }
    __syncthreads();

    for (int it = 0; it < 1000; ++it) {
        if (t < 100) {                      // f_i from old g (rows)
            float m = -__builtin_inff(), s = 0.f;
            const float* Cr = &C[t * 100];
            for (int j = 0; j < 100; ++j) {
                float x = (g[j] - Cr[j]) * invreg;
                if (x > m) { s = s * __expf(m - x) + 1.f; m = x; }
                else       { s += __expf(x - m); }
            }
            f[t] = reg * (la[t] - (m + __logf(s)));
        }
        __syncthreads();
        if (t < 100) {                      // g_j from new f (columns)
            float m = -__builtin_inff(), s = 0.f;
            for (int i2 = 0; i2 < 100; ++i2) {
                float x = (f[i2] - C[i2 * 100 + t]) * invreg;
                if (x > m) { s = s * __expf(m - x) + 1.f; m = x; }
                else       { s += __expf(x - m); }
            }
            g[t] = reg * (lb[t] - (m + __logf(s)));
        }
        __syncthreads();
    }

    float part = 0.f;
    for (int p = t; p < 10000; p += 128) {
        int i2 = p / 100, j = p % 100;
        part += __expf((f[i2] + g[j] - C[p]) * invreg) * C[p];
    }
    red[t] = part; __syncthreads();
    for (int s = 64; s > 0; s >>= 1) { if (t < s) red[t] += red[t + s]; __syncthreads(); }
    if (t == 0) atomicAdd(sink_acc, red[0]);
}

// ---------------------------------------------------------------------------
__global__ void zero_accum(float* a) { if (threadIdx.x < 8) a[threadIdx.x] = 0.f; }

__global__ void finalize_loss(const float* __restrict__ a, float* __restrict__ out)
{
    // a: [0]=kld_en [1]=recon_en [2]=kld_cn [3]=recon_cn [4]=sinkhorn
    const float invB = 1.f / 2048.f;
    out[0] = (a[0] + a[1]) * invB + (a[2] + a[3]) * invB + 0.5f * a[4];
}

// ---------------------------------------------------------------------------
extern "C" void kernel_launch(void* const* d_in, const int* in_sizes, int n_in,
                              void* d_out, int out_size, void* d_ws, size_t ws_size,
                              hipStream_t stream)
{
    (void)in_sizes; (void)n_in; (void)out_size; (void)ws_size;
    const int B = 2048, V = 20000, H = 200, K = 100, D = 300;

    const float* x_en   = (const float*)d_in[0];
    const float* x_cn   = (const float*)d_in[1];
    const float* eps_en = (const float*)d_in[2];
    const float* eps_cn = (const float*)d_in[3];
    const float* W11_en = (const float*)d_in[4];
    const float* b11_en = (const float*)d_in[5];
    const float* W12_en = (const float*)d_in[6];
    const float* b12_en = (const float*)d_in[7];
    const float* W21_en = (const float*)d_in[8];
    const float* b21_en = (const float*)d_in[9];
    const float* W22_en = (const float*)d_in[10];
    const float* b22_en = (const float*)d_in[11];
    const float* W11_cn = (const float*)d_in[12];
    const float* b11_cn = (const float*)d_in[13];
    const float* W12_cn = (const float*)d_in[14];
    const float* b12_cn = (const float*)d_in[15];
    const float* W21_cn = (const float*)d_in[16];
    const float* b21_cn = (const float*)d_in[17];
    const float* W22_cn = (const float*)d_in[18];
    const float* b22_cn = (const float*)d_in[19];
    const float* phi_en = (const float*)d_in[20];
    const float* phi_cn = (const float*)d_in[21];
    const float* BWE_en = (const float*)d_in[22];
    const float* BWE_cn = (const float*)d_in[23];

    // Workspace layout (floats). Total ~230 MB.
    float* ws     = (float*)d_ws;
    float* Y      = ws;                        // 2048*20000
    float* e1     = Y      + (size_t)B * V;    // 2048*200
    float* e2     = e1     + (size_t)B * H;    // 2048*200
    float* mu_pre = e2     + (size_t)B * H;    // 2048*100
    float* lv_pre = mu_pre + (size_t)B * K;    // 2048*100
    float* theta  = lv_pre + (size_t)B * K;    // 2048*100
    float* bn4    = theta  + (size_t)B * K;    // 4*100
    float* cmean  = bn4    + 400;              // 20000
    float* cvar   = cmean  + V;                // 20000
    float* en_n   = cvar   + V;                // 20000*300
    float* cn_n   = en_n   + (size_t)V * D;    // 20000*300
    float* tscr   = cn_n   + (size_t)V * D;    // 100*20000
    float* v_cn   = tscr   + (size_t)K * V;    // 100*100
    float* v_en   = v_cn   + 10000;            // 100*100
    int*   i_cn   = (int*)(v_en + 10000);      // 100*100
    int*   i_en   = i_cn   + 10000;            // 100*100
    float* Cm     = (float*)(i_en + 10000);    // 100*100*100
    float* acc    = Cm     + 1000000;          // 8

    float* bn_mm = bn4;        // mu batchnorm mean
    float* bn_mv = bn4 + 100;  // mu batchnorm var
    float* bn_lm = bn4 + 200;  // logvar batchnorm mean
    float* bn_lv = bn4 + 300;  // logvar batchnorm var

    zero_accum<<<1, 32, 0, stream>>>(acc);

    const float* xs[2]   = {x_en, x_cn};
    const float* epss[2] = {eps_en, eps_cn};
    const float* W11s[2] = {W11_en, W11_cn};
    const float* b11s[2] = {b11_en, b11_cn};
    const float* W12s[2] = {W12_en, W12_cn};
    const float* b12s[2] = {b12_en, b12_cn};
    const float* W21s[2] = {W21_en, W21_cn};
    const float* b21s[2] = {b21_en, b21_cn};
    const float* W22s[2] = {W22_en, W22_cn};
    const float* b22s[2] = {b22_en, b22_cn};
    const float* phis[2] = {phi_en, phi_cn};

    for (int l = 0; l < 2; ++l) {
        // Encoder layer 1: e1 = softplus(x @ W11 + b11)  [2048 x 200], K=20000
        gemm_bf16_wmma<<<dim3(B / 64, (H + 63) / 64), 256, 0, stream>>>(
            xs[l], W11s[l], b11s[l], e1, B, H, V, 1);
        // Encoder layer 2: e2 = softplus(e1 @ W12 + b12)
        gemm_bf16_wmma<<<dim3(B / 64, (H + 63) / 64), 256, 0, stream>>>(
            e1, W12s[l], b12s[l], e2, B, H, H, 1);
        // mu_pre / logvar_pre
        gemm_bf16_wmma<<<dim3(B / 64, (K + 63) / 64), 256, 0, stream>>>(
            e2, W21s[l], b21s[l], mu_pre, B, K, H, 0);
        gemm_bf16_wmma<<<dim3(B / 64, (K + 63) / 64), 256, 0, stream>>>(
            e2, W22s[l], b22s[l], lv_pre, B, K, H, 0);
        // Batchnorm stats (over batch dim)
        colstats<<<(K + 255) / 256, 256, 0, stream>>>(mu_pre, bn_mm, bn_mv, B, K);
        colstats<<<(K + 255) / 256, 256, 0, stream>>>(lv_pre, bn_lm, bn_lv, B, K);
        // BN + reparameterize + softmax(theta) + KL accumulation
        bn_reparam_theta<<<B, 128, 0, stream>>>(mu_pre, lv_pre, bn_mm, bn_mv,
                                                bn_lm, bn_lv, epss[l], theta,
                                                acc + l * 2);
        // Decode: Y = theta @ phi   [2048 x 20000]
        gemm_bf16_wmma<<<dim3(B / 64, (V + 63) / 64), 256, 0, stream>>>(
            theta, phis[l], nullptr, Y, B, V, K, 0);
        // Column batchnorm stats over Y, then row softmax + recon loss
        colstats<<<(V + 255) / 256, 256, 0, stream>>>(Y, cmean, cvar, B, V);
        row_softmax_loss<<<B, 256, 0, stream>>>(Y, cmean, cvar, xs[l],
                                                acc + l * 2 + 1, V);
    }

    // Sinkhorn path (detached): normalized embeddings, per-topic top-100,
    // 100x100 cost tiles, 1000-iteration log-domain Sinkhorn.
    norm_rows<<<V, 64, 0, stream>>>(BWE_en, en_n, D);
    norm_rows<<<V, 64, 0, stream>>>(BWE_cn, cn_n, D);
    topk100<<<K, 256, 0, stream>>>(phi_cn, tscr, v_cn, i_cn, V);
    topk100<<<K, 256, 0, stream>>>(phi_en, tscr, v_en, i_en, V);
    build_cost<<<K, 256, 0, stream>>>(en_n, cn_n, i_cn, i_en, Cm);
    sinkhorn100<<<K, 128, 0, stream>>>(Cm, v_cn, v_en, acc + 4);

    finalize_loss<<<1, 1, 0, stream>>>(acc, (float*)d_out);
}
